// Decoder_24275155157483
// MI455X (gfx1250) — matrix operand order repaired and checked by reading, not compiled, not run on hardware
//
#include <hip/hip_runtime.h>

typedef __bf16 bf16_t;
typedef __attribute__((ext_vector_type(16))) __bf16 v16bf;
typedef __attribute__((ext_vector_type(8)))  float    v8f;

#define Bsz    4
#define SEQ    512
#define DIM    1024
#define FFD    4096
#define NH     16
#define HDm    64
#define NLAYER 6
#define VOCAB  32000

__device__ __forceinline__ bf16_t f2bf(float f) { return (bf16_t)f; }

// Low 32 bits of a generic pointer to a __shared__ object = LDS byte offset
// (ISA 10.2: LDS aperture truncates addr[31:0]).
__device__ __forceinline__ unsigned lds_off(const void* p) {
  return (unsigned)(size_t)p;
}

// gfx1250 async global->LDS 128b copy (ASYNCcnt-tracked, no VGPR round-trip).
__device__ __forceinline__ void async_load_b128(unsigned lds_addr, const void* gaddr) {
  asm volatile("global_load_async_to_lds_b128 %0, %1, off"
               :: "v"(lds_addr), "v"((unsigned long long)(size_t)gaddr)
               : "memory");
}
__device__ __forceinline__ void wait_async0() {
  asm volatile("s_wait_asynccnt 0" ::: "memory");
}

// ---------------------------------------------------------------------------
// C[M,N] = A[M,K] (f32) @ W[N,K]^T (bf16) + bias, optional ReLU.
// Output: f32 to C (Cbf==null) or bf16 to Cbf (for attention Q/K/V).
// Block tile 128x128, BK=32, 256 threads = 8 waves, wave tile 64x32.
// Double-buffered LDS; weight tiles staged via async-to-LDS, overlapped with
// the 8 WMMAs of the current tile.
// ---------------------------------------------------------------------------
__global__ __launch_bounds__(256) void gemm_bf16_kernel(
    const float* __restrict__ A, const bf16_t* __restrict__ W,
    const float* __restrict__ bias, float* __restrict__ C,
    bf16_t* __restrict__ Cbf, int M, int N, int K, int relu)
{
  __shared__ __align__(16) bf16_t ldsA[2][128 * 32];
  __shared__ __align__(16) bf16_t ldsB[2][128 * 32];

  const int tid    = threadIdx.x;
  const int lane   = tid & 31;
  const int wave   = tid >> 5;
  const int laneLo = lane & 15;
  const int laneHi = lane >> 4;
  const int m0 = blockIdx.y * 128;
  const int n0 = blockIdx.x * 128;
  const int wm = (wave & 1) * 64;
  const int wn = (wave >> 1) * 32;
  const int lrow = tid >> 1;
  const int loff = (tid & 1) * 16;

  v8f acc[4][2];
  #pragma unroll
  for (int mi = 0; mi < 4; mi++)
    #pragma unroll
    for (int ni = 0; ni < 2; ni++)
      #pragma unroll
      for (int r = 0; r < 8; r++) acc[mi][ni][r] = 0.0f;

  // ---- prologue: stage tile 0 into buffer 0 ----
  {
    const float* ap = A + (size_t)(m0 + lrow) * K + loff;
    union { bf16_t h[16]; float4 q[2]; } ua;
    #pragma unroll
    for (int i = 0; i < 16; i++) ua.h[i] = f2bf(ap[i]);
    *(float4*)(&ldsA[0][lrow * 32 + loff])     = ua.q[0];
    *(float4*)(&ldsA[0][lrow * 32 + loff + 8]) = ua.q[1];
    const bf16_t* wp = W + (size_t)(n0 + lrow) * K + loff;
    async_load_b128(lds_off(&ldsB[0][lrow * 32 + loff]),     wp);
    async_load_b128(lds_off(&ldsB[0][lrow * 32 + loff + 8]), wp + 8);
  }
  wait_async0();
  __syncthreads();

  for (int kt = 0; kt < K; kt += 32) {
    const int cur = (kt >> 5) & 1;
    // ---- stage next tile into the other buffer (overlaps with WMMAs) ----
    if (kt + 32 < K) {
      const int nb = cur ^ 1;
      const float* ap = A + (size_t)(m0 + lrow) * K + (kt + 32) + loff;
      union { bf16_t h[16]; float4 q[2]; } ua;
      #pragma unroll
      for (int i = 0; i < 16; i++) ua.h[i] = f2bf(ap[i]);
      *(float4*)(&ldsA[nb][lrow * 32 + loff])     = ua.q[0];
      *(float4*)(&ldsA[nb][lrow * 32 + loff + 8]) = ua.q[1];
      const bf16_t* wp = W + (size_t)(n0 + lrow) * K + (kt + 32) + loff;
      async_load_b128(lds_off(&ldsB[nb][lrow * 32 + loff]),     wp);
      async_load_b128(lds_off(&ldsB[nb][lrow * 32 + loff + 8]), wp + 8);
    }

    // ---- compute on current buffer ----
    // A frag: lane = M (0..15), laneHi selects K {0..7,16..23} / {8..15,24..31}
    v16bf af[4], bfr[2];
    const int ka = laneHi * 8;
    const int kb = laneHi * 8;
    #pragma unroll
    for (int mi = 0; mi < 4; mi++) {
      union { v16bf v; float4 q[2]; } u;
      const bf16_t* p = &ldsA[cur][(size_t)(wm + mi * 16 + laneLo) * 32];
      u.q[0] = *(const float4*)(p + ka);
      u.q[1] = *(const float4*)(p + ka + 16);
      af[mi] = u.v;
    }
    // B frag: lane = N (0..15), laneHi selects K {0..7,16..23} / {8..15,24..31}
    #pragma unroll
    for (int ni = 0; ni < 2; ni++) {
      union { v16bf v; float4 q[2]; } u;
      const bf16_t* p = &ldsB[cur][(size_t)(wn + ni * 16 + laneLo) * 32];
      u.q[0] = *(const float4*)(p + kb);
      u.q[1] = *(const float4*)(p + kb + 16);
      bfr[ni] = u.v;
    }
    #pragma unroll
    for (int mi = 0; mi < 4; mi++)
      #pragma unroll
      for (int ni = 0; ni < 2; ni++)
        acc[mi][ni] = __builtin_amdgcn_wmma_f32_16x16x32_bf16(
            false, af[mi], false, bfr[ni], (short)0, acc[mi][ni], false, false);

    wait_async0();      // next tile's async B stores into LDS complete
    __syncthreads();    // all waves done reading cur / writing next
  }

  // Epilogue: C layout per ISA: M = r + 8*laneHi, N = laneLo (branch hoisted)
  if (Cbf) {
    #pragma unroll
    for (int mi = 0; mi < 4; mi++) {
      #pragma unroll
      for (int ni = 0; ni < 2; ni++) {
        const int col = n0 + wn + ni * 16 + laneLo;
        const float bb = bias[col];
        bf16_t* cp = Cbf + (size_t)(m0 + wm + mi * 16 + 8 * laneHi) * N + col;
        #pragma unroll
        for (int r = 0; r < 8; r++)
          cp[(size_t)r * N] = f2bf(acc[mi][ni][r] + bb);
      }
    }
  } else {
    #pragma unroll
    for (int mi = 0; mi < 4; mi++) {
      #pragma unroll
      for (int ni = 0; ni < 2; ni++) {
        const int col = n0 + wn + ni * 16 + laneLo;
        const float bb = bias[col];
        float* cp = C + (size_t)(m0 + wm + mi * 16 + 8 * laneHi) * N + col;
        #pragma unroll
        for (int r = 0; r < 8; r++) {
          float v = acc[mi][ni][r] + bb;
          if (relu) v = fmaxf(v, 0.0f);
          cp[(size_t)r * N] = v;
        }
      }
    }
  }
}

// ---------------------------------------------------------------------------
// Attention: one wave per (b, h, 16-query tile). Sq = Sk = 512, HD = 64.
// Q/K/V are bf16 [B,S,H,HD]. Scores in f32 LDS; probabilities re-staged as
// bf16 in LDS so PV A-frags are two ds_load_b128 each.
// ---------------------------------------------------------------------------
__global__ __launch_bounds__(32) void attn_kernel(
    const bf16_t* __restrict__ Q, const bf16_t* __restrict__ Kt,
    const bf16_t* __restrict__ Vt, float* __restrict__ O, int causal)
{
  __shared__ __align__(16) float  s_sc[16 * 512];
  __shared__ __align__(16) bf16_t s_p [16 * 512];

  const int lane   = threadIdx.x;
  const int laneLo = lane & 15;
  const int laneHi = lane >> 4;
  const int wid = blockIdx.x;
  const int qt = wid & 31;
  const int bh = wid >> 5;
  const int h  = bh & (NH - 1);
  const int b  = bh >> 4;

  // Q fragments (16x64 -> two 16x32 A frags), vector loads
  const bf16_t* Qb = Q + ((size_t)(b * SEQ + qt * 16 + laneLo) * NH + h) * HDm;
  v16bf aq[2];
  #pragma unroll
  for (int j = 0; j < 2; j++) {
    union { v16bf v; float4 q[2]; } u;
    const int kb = j * 32 + laneHi * 8;
    u.q[0] = *(const float4*)(Qb + kb);
    u.q[1] = *(const float4*)(Qb + kb + 16);
    aq[j] = u.v;
  }

  const float scale = 0.125f;   // 1/sqrt(64)
  const float NEG = -__builtin_inff();

  // Scores: 32 key tiles of 16, K = HD = 64 via two WMMAs
  #pragma unroll 1
  for (int nt = 0; nt < 32; nt++) {
    v8f acc = {0.f, 0.f, 0.f, 0.f, 0.f, 0.f, 0.f, 0.f};
    const bf16_t* Kp = Kt + ((size_t)(b * SEQ + nt * 16 + laneLo) * NH + h) * HDm;
    #pragma unroll
    for (int j = 0; j < 2; j++) {
      union { v16bf v; float4 q[2]; } u;
      const int kb = j * 32 + laneHi * 8;
      u.q[0] = *(const float4*)(Kp + kb);
      u.q[1] = *(const float4*)(Kp + kb + 16);
      acc = __builtin_amdgcn_wmma_f32_16x16x32_bf16(false, aq[j], false, u.v,
                                                    (short)0, acc, false, false);
    }
    const int key = nt * 16 + laneLo;
    #pragma unroll
    for (int r = 0; r < 8; r++) {
      const int qrow  = r + 8 * laneHi;
      const int query = qt * 16 + qrow;
      const float val = acc[r] * scale;
      s_sc[qrow * 512 + key] = (causal && key > query) ? NEG : val;
    }
  }
  __syncthreads();

  // Row softmax: 2 lanes per row, 256 keys each; write bf16 probabilities.
  {
    const int row = lane >> 1, half = lane & 1;
    float*  sr = s_sc + row * 512 + half * 256;
    bf16_t* pr = s_p  + row * 512 + half * 256;
    float mx = -3.0e38f;
    #pragma unroll 1
    for (int j = 0; j < 256; j++) mx = fmaxf(mx, sr[j]);
    mx = fmaxf(mx, __shfl_xor(mx, 1, 32));
    float sum = 0.0f;
    #pragma unroll 1
    for (int j = 0; j < 256; j++) { float e = __expf(sr[j] - mx); sr[j] = e; sum += e; }
    sum += __shfl_xor(sum, 1, 32);
    const float inv = 1.0f / sum;
    #pragma unroll 1
    for (int j = 0; j < 256; j++) pr[j] = f2bf(sr[j] * inv);
  }
  __syncthreads();

  // Out = P @ V : four 16-wide HD subtiles, K loop over 512 keys in steps of 32
  const bf16_t* Vb = Vt + ((size_t)b * SEQ * NH + h) * HDm;
  float* Ob = O + ((size_t)(b * SEQ + qt * 16) * NH + h) * HDm;
  #pragma unroll 1
  for (int nsub = 0; nsub < 4; nsub++) {
    v8f acc = {0.f, 0.f, 0.f, 0.f, 0.f, 0.f, 0.f, 0.f};
    #pragma unroll 1
    for (int kt2 = 0; kt2 < 16; kt2++) {
      // A frag: bf16 probabilities, two ds_load_b128
      union { v16bf v; float4 q[2]; } ua;
      const bf16_t* pp = s_p + laneLo * 512 + kt2 * 32 + laneHi * 8;
      ua.q[0] = *(const float4*)(pp);
      ua.q[1] = *(const float4*)(pp + 16);
      // B frag: V gather (key-strided), bf16
      v16bf bv;
      const int kk0 = kt2 * 32 + laneHi * 8;
      const int hd  = nsub * 16 + laneLo;
      #pragma unroll
      for (int e = 0; e < 16; e++) bv[e] = Vb[(size_t)(kk0 + e + (e >= 8 ? 8 : 0)) * DIM + hd];
      acc = __builtin_amdgcn_wmma_f32_16x16x32_bf16(false, ua.v, false, bv,
                                                    (short)0, acc, false, false);
    }
    #pragma unroll
    for (int r = 0; r < 8; r++)
      Ob[(size_t)(r + 8 * laneHi) * DIM + nsub * 16 + laneLo] = acc[r];
  }
}

// ---------------------------------------------------------------------------
// y = LayerNorm(x + r) * g + b  over D=1024, one row per block (256 threads)
// ---------------------------------------------------------------------------
__global__ __launch_bounds__(256) void ln_add_kernel(
    const float* __restrict__ x, const float* __restrict__ r,
    const float* __restrict__ g, const float* __restrict__ bb,
    float* __restrict__ y)
{
  __shared__ float red[16];
  __shared__ float s_mu, s_inv;
  const int row = blockIdx.x, tid = threadIdx.x;
  float4 xv = ((const float4*)(x + (size_t)row * DIM))[tid];
  float4 rv = ((const float4*)(r + (size_t)row * DIM))[tid];
  float v0 = xv.x + rv.x, v1 = xv.y + rv.y, v2 = xv.z + rv.z, v3 = xv.w + rv.w;
  float s = v0 + v1 + v2 + v3;
  float q = v0 * v0 + v1 * v1 + v2 * v2 + v3 * v3;
  #pragma unroll
  for (int off = 16; off > 0; off >>= 1) {
    s += __shfl_xor(s, off, 32);
    q += __shfl_xor(q, off, 32);
  }
  const int wave = tid >> 5, lane = tid & 31;
  if (!lane) { red[wave] = s; red[8 + wave] = q; }
  __syncthreads();
  if (tid == 0) {
    float S = 0.f, Qs = 0.f;
    for (int w = 0; w < 8; w++) { S += red[w]; Qs += red[8 + w]; }
    float mu = S / (float)DIM;
    float var = Qs / (float)DIM - mu * mu;
    s_mu = mu; s_inv = rsqrtf(var + 1e-5f);
  }
  __syncthreads();
  const float mu = s_mu, inv = s_inv;
  float4 gv = ((const float4*)g)[tid];
  float4 bv = ((const float4*)bb)[tid];
  float4 o;
  o.x = (v0 - mu) * inv * gv.x + bv.x;
  o.y = (v1 - mu) * inv * gv.y + bv.y;
  o.z = (v2 - mu) * inv * gv.z + bv.z;
  o.w = (v3 - mu) * inv * gv.w + bv.w;
  ((float4*)(y + (size_t)row * DIM))[tid] = o;
}

// ---------------------------------------------------------------------------
// x[b,s,:] = W_e[text[b,s],:] + sinusoidal positional encoding(s,:)
// ---------------------------------------------------------------------------
__global__ __launch_bounds__(256) void embed_pe_kernel(
    const int* __restrict__ text, const float* __restrict__ We,
    float* __restrict__ x)
{
  const int row = blockIdx.x;          // b*SEQ + s
  const int s = row & (SEQ - 1);
  const int tok = text[row];
  const int tid = threadIdx.x;
  float4 wv = ((const float4*)(We + (size_t)tok * DIM))[tid];
  float out[4] = {wv.x, wv.y, wv.z, wv.w};
  #pragma unroll
  for (int i = 0; i < 4; i++) {
    int d = tid * 4 + i;
    float freq = __expf(-9.210340371976184f * (float)(d & ~1) / (float)DIM);
    float arg = (float)s * freq;
    out[i] += (d & 1) ? __cosf(arg) : __sinf(arg);
  }
  float4 o = make_float4(out[0], out[1], out[2], out[3]);
  ((float4*)(x + (size_t)row * DIM))[tid] = o;
}

// f32 -> bf16 weight conversion (4 elements/thread)
__global__ __launch_bounds__(256) void cvt_kernel(
    const float* __restrict__ s, bf16_t* __restrict__ d, long long n)
{
  long long i = ((long long)blockIdx.x * 256 + threadIdx.x) * 4;
  if (i >= n) return;
  float4 v = *(const float4*)(s + i);
  d[i + 0] = f2bf(v.x); d[i + 1] = f2bf(v.y);
  d[i + 2] = f2bf(v.z); d[i + 3] = f2bf(v.w);
}

// In-place softmax over batch axis (axis 0) of logits [B, S, V]
__global__ __launch_bounds__(256) void softmax_b_kernel(float* __restrict__ out)
{
  const size_t SV = (size_t)SEQ * VOCAB;
  size_t idx = (size_t)blockIdx.x * 256 + threadIdx.x;
  if (idx >= SV) return;
  float a0 = out[idx], a1 = out[idx + SV], a2 = out[idx + 2 * SV], a3 = out[idx + 3 * SV];
  float m = fmaxf(fmaxf(a0, a1), fmaxf(a2, a3));
  float e0 = __expf(a0 - m), e1 = __expf(a1 - m), e2 = __expf(a2 - m), e3 = __expf(a3 - m);
  float inv = 1.0f / (e0 + e1 + e2 + e3);
  out[idx] = e0 * inv; out[idx + SV] = e1 * inv;
  out[idx + 2 * SV] = e2 * inv; out[idx + 3 * SV] = e3 * inv;
}

// ---------------------------------------------------------------------------
extern "C" void kernel_launch(void* const* d_in, const int* in_sizes, int n_in,
                              void* d_out, int out_size, void* d_ws, size_t ws_size,
                              hipStream_t stream) {
  const int*   text = (const int*)d_in[0];
  const float* enc  = (const float*)d_in[1];
  const float* We   = (const float*)d_in[2];
  const float* bout = (const float*)d_in[3];
  const float* lng  = (const float*)d_in[4];
  const float* lnb  = (const float*)d_in[5];
  const float* W1   = (const float*)d_in[6];
  const float* b1   = (const float*)d_in[7];
  const float* W2   = (const float*)d_in[8];
  const float* b2   = (const float*)d_in[9];
  const float* mWq = (const float*)d_in[10], *mbq = (const float*)d_in[11];
  const float* mWk = (const float*)d_in[12], *mbk = (const float*)d_in[13];
  const float* mWv = (const float*)d_in[14], *mbv = (const float*)d_in[15];
  const float* mWo = (const float*)d_in[16], *mbo = (const float*)d_in[17];
  const float* cWq = (const float*)d_in[18], *cbq = (const float*)d_in[19];
  const float* cWk = (const float*)d_in[20], *cbk = (const float*)d_in[21];
  const float* cWv = (const float*)d_in[22], *cbv = (const float*)d_in[23];
  const float* cWo = (const float*)d_in[24], *cbo = (const float*)d_in[25];

  char* wsp = (char*)d_ws;
  size_t off = 0;
  auto alloc = [&](size_t bytes) -> void* {
    void* p = wsp + off;
    off = (off + bytes + 255) & ~(size_t)255;
    return p;
  };

  const size_t M  = (size_t)Bsz * SEQ;           // 2048
  const size_t MD = M * DIM;

  // bf16 weight copies
  bf16_t* We_bf = (bf16_t*)alloc((size_t)VOCAB * DIM * 2);
  bf16_t* W1_bf = (bf16_t*)alloc((size_t)FFD * DIM * 2);
  bf16_t* W2_bf = (bf16_t*)alloc((size_t)DIM * FFD * 2);
  bf16_t* aw[8];
  for (int i = 0; i < 8; i++) aw[i] = (bf16_t*)alloc((size_t)DIM * DIM * 2);
  bf16_t *mWq_bf = aw[0], *mWk_bf = aw[1], *mWv_bf = aw[2], *mWo_bf = aw[3];
  bf16_t *cWq_bf = aw[4], *cWk_bf = aw[5], *cWv_bf = aw[6], *cWo_bf = aw[7];

  // activation buffers
  float*  x     = (float*)alloc(MD * 4);
  float*  bufa  = (float*)alloc(MD * 4);
  float*  bufc  = (float*)alloc(MD * 4);
  float*  attn  = (float*)alloc(MD * 4);
  float*  proj  = (float*)alloc(MD * 4);
  float*  ffo   = (float*)alloc(MD * 4);
  float*  ffh   = (float*)alloc(M * FFD * 4);
  bf16_t* qbf   = (bf16_t*)alloc(MD * 2);
  bf16_t* kbf   = (bf16_t*)alloc(MD * 2);
  bf16_t* vbf   = (bf16_t*)alloc(MD * 2);
  bf16_t* kencb = (bf16_t*)alloc(MD * 2);
  bf16_t* vencb = (bf16_t*)alloc(MD * 2);

  auto cvt = [&](const float* s, bf16_t* d, size_t n) {
    cvt_kernel<<<(unsigned)((n + 1023) / 1024), 256, 0, stream>>>(s, d, (long long)n);
  };
  auto gemm = [&](const float* A, const bf16_t* Wm, const float* bias,
                  float* C, bf16_t* Cb, int Mm, int Nn, int Kk, int relu) {
    dim3 g((unsigned)(Nn / 128), (unsigned)(Mm / 128));
    gemm_bf16_kernel<<<g, 256, 0, stream>>>(A, Wm, bias, C, Cb, Mm, Nn, Kk, relu);
  };

  // 1) weights -> bf16 (once per launch; deterministic)
  cvt(We, We_bf, (size_t)VOCAB * DIM);
  cvt(W1, W1_bf, (size_t)FFD * DIM);
  cvt(W2, W2_bf, (size_t)DIM * FFD);
  cvt(mWq, mWq_bf, (size_t)DIM * DIM); cvt(mWk, mWk_bf, (size_t)DIM * DIM);
  cvt(mWv, mWv_bf, (size_t)DIM * DIM); cvt(mWo, mWo_bf, (size_t)DIM * DIM);
  cvt(cWq, cWq_bf, (size_t)DIM * DIM); cvt(cWk, cWk_bf, (size_t)DIM * DIM);
  cvt(cWv, cWv_bf, (size_t)DIM * DIM); cvt(cWo, cWo_bf, (size_t)DIM * DIM);

  // 2) embedding + positional encoding
  embed_pe_kernel<<<(unsigned)M, 256, 0, stream>>>(text, We, x);

  // 3) cross-attention K/V from encoder output (loop-invariant, bf16 out)
  gemm(enc, cWk_bf, cbk, nullptr, kencb, (int)M, DIM, DIM, 0);
  gemm(enc, cWv_bf, cbv, nullptr, vencb, (int)M, DIM, DIM, 0);

  const unsigned attn_grid = Bsz * NH * (SEQ / 16);   // 2048 wave-workgroups

  // 4) decoder layers (weights shared across layers, as in reference)
  for (int l = 0; l < NLAYER; l++) {
    // masked self-attention (Q/K/V written directly as bf16)
    gemm(x, mWq_bf, mbq, nullptr, qbf, (int)M, DIM, DIM, 0);
    gemm(x, mWk_bf, mbk, nullptr, kbf, (int)M, DIM, DIM, 0);
    gemm(x, mWv_bf, mbv, nullptr, vbf, (int)M, DIM, DIM, 0);
    attn_kernel<<<attn_grid, 32, 0, stream>>>(qbf, kbf, vbf, attn, 1);
    gemm(attn, mWo_bf, mbo, proj, nullptr, (int)M, DIM, DIM, 0);
    ln_add_kernel<<<(unsigned)M, 256, 0, stream>>>(x, proj, lng, lnb, bufa);

    // cross-attention
    gemm(bufa, cWq_bf, cbq, nullptr, qbf, (int)M, DIM, DIM, 0);
    attn_kernel<<<attn_grid, 32, 0, stream>>>(qbf, kencb, vencb, attn, 0);
    gemm(attn, cWo_bf, cbo, proj, nullptr, (int)M, DIM, DIM, 0);
    ln_add_kernel<<<(unsigned)M, 256, 0, stream>>>(proj, bufa, lng, lnb, bufc);

    // FFN
    gemm(bufc, W1_bf, b1, ffh, nullptr, (int)M, FFD, DIM, 1);   // ReLU epilogue
    gemm(ffh, W2_bf, b2, ffo, nullptr, (int)M, DIM, FFD, 0);
    ln_add_kernel<<<(unsigned)M, 256, 0, stream>>>(bufc, ffo, lng, lnb, x);
  }

  // 5) tied-embedding logits straight into d_out, then softmax over batch axis
  gemm(x, We_bf, bout, (float*)d_out, nullptr, (int)M, VOCAB, DIM, 0);
  const size_t SV = (size_t)SEQ * VOCAB;
  softmax_b_kernel<<<(unsigned)((SV + 255) / 256), 256, 0, stream>>>((float*)d_out);
}
